// NystromformerSelfAttention_14499809591787
// MI455X (gfx1250) — matrix operand order repaired and checked by reading, not compiled, not run on hardware
//
#include <hip/hip_runtime.h>
#include <stdint.h>

// ---------------------------------------------------------------------------
// Nystromformer self-attention forward, MI455X (gfx1250), wave32 WMMA bf16.
// B=4 S=4096 HID=1024 H=16 D=64 L=64 K=65
// ---------------------------------------------------------------------------

static constexpr int Bc   = 4;
static constexpr int Sc   = 4096;
static constexpr int HIDc = 1024;
static constexpr int Hc   = 16;
static constexpr int Dc   = 64;
static constexpr int Lc   = 64;
static constexpr int Kc   = 65;
static constexpr int BH   = Bc * Hc;          // 64
static constexpr int ROWS = Bc * Sc;          // 16384

typedef __bf16 bf16_t;
typedef bf16_t bf16x16 __attribute__((ext_vector_type(16)));
typedef float  f32x8   __attribute__((ext_vector_type(8)));

__device__ __forceinline__ uint16_t f2bf(float f) {
  union { float f; uint32_t u; } c; c.f = f;
  uint32_t u = c.u;
  u += 0x7fffu + ((u >> 16) & 1u);   // round-to-nearest-even
  return (uint16_t)(u >> 16);
}
__device__ __forceinline__ float bf2f(uint16_t h) {
  union { uint32_t u; float f; } c; c.u = ((uint32_t)h) << 16;
  return c.f;
}

union FragU { uint4 q[2]; bf16x16 v; };

// A fragment: A row-major [M x K] bf16. lane m = m0 + lane%16, hi = lane/16.
// vector elems e<8 -> K = k0 + 8*hi + e ; e>=8 -> K = k0 + 16 + 8*hi + (e-8).
__device__ __forceinline__ bf16x16 load_a_frag(const uint16_t* __restrict__ base,
                                               int lda, int m0, int k0) {
  const int lane = threadIdx.x & 31;
  const int hi   = lane >> 4;
  const uint16_t* row = base + (size_t)(m0 + (lane & 15)) * lda + k0;
  FragU f;
  f.q[0] = *(const uint4*)(row + 8 * hi);
  f.q[1] = *(const uint4*)(row + 16 + 8 * hi);
  return f.v;
}

// B fragment from N-major storage Bs[n][k] (row n contiguous in k).
// lane n = n0 + lane%16, hi = lane/16; e<8 -> K = k0 + 8*hi + e ; e>=8 -> K = k0 + 16 + 8*hi + (e-8).
__device__ __forceinline__ bf16x16 load_b_frag(const uint16_t* __restrict__ base,
                                               int ldb, int n0, int k0) {
  const int lane = threadIdx.x & 31;
  const int hi   = lane >> 4;
  const uint16_t* row = base + (size_t)(n0 + (lane & 15)) * ldb + k0;
  FragU f;
  f.q[0] = *(const uint4*)(row + 8 * hi);
  f.q[1] = *(const uint4*)(row + 16 + 8 * hi);
  return f.v;
}

__device__ __forceinline__ f32x8 wmma_bf(bf16x16 a, bf16x16 b, f32x8 c) {
  return __builtin_amdgcn_wmma_f32_16x16x32_bf16(false, a, false, b,
                                                 (short)0, c, false, false);
}

// C/D layout helper: element e of lane -> row m0 + e + 8*(lane/16), col n0+16t+lane%16.

// ---------------------------------------------------------------------------
// 1) fp32 -> bf16 conversion
// ---------------------------------------------------------------------------
__global__ void k_f32_to_bf16(const float* __restrict__ in,
                              uint16_t* __restrict__ out, int n) {
  int i = blockIdx.x * blockDim.x + threadIdx.x;
  int stride = gridDim.x * blockDim.x;
  for (; i < n; i += stride) out[i] = f2bf(in[i]);
}

// ---------------------------------------------------------------------------
// 2) QKV GEMM: out = X @ W^T + b, write head layout [B,H,S,D].
//    z = 0:q (scaled bf16), 1:k (scaled bf16), 2:v (f32 + bf16 transposed)
// ---------------------------------------------------------------------------
__global__ __launch_bounds__(256)
void k_qkv_gemm(const uint16_t* __restrict__ X,
                const uint16_t* __restrict__ Wq, const uint16_t* __restrict__ Wk,
                const uint16_t* __restrict__ Wv,
                const float* __restrict__ bq, const float* __restrict__ bk,
                const float* __restrict__ bv,
                uint16_t* __restrict__ qbf, uint16_t* __restrict__ kbf,
                float* __restrict__ vf, uint16_t* __restrict__ vT) {
  const int which = blockIdx.z;
  const uint16_t* W = (which == 0) ? Wq : (which == 1) ? Wk : Wv;
  const float* bias = (which == 0) ? bq : (which == 1) ? bk : bv;
  const int wave = threadIdx.x >> 5;
  const int lane = threadIdx.x & 31;
  const int m0 = blockIdx.x * 16;
  const int n0 = blockIdx.y * 512 + wave * 64;

  f32x8 acc[4] = {};
  for (int k0 = 0; k0 < HIDc; k0 += 32) {
    bf16x16 a = load_a_frag(X, HIDc, m0, k0);
#pragma unroll
    for (int t = 0; t < 4; ++t) {
      bf16x16 b = load_b_frag(W, HIDc, n0 + 16 * t, k0);
      acc[t] = wmma_bf(a, b, acc[t]);
    }
  }

  const float scale = 0.35355339059327373f;  // D^-0.25
  const int hi = lane >> 4;
#pragma unroll
  for (int t = 0; t < 4; ++t) {
    const int c = n0 + 16 * t + (lane & 15);
    const int h = c >> 6, d = c & 63;
    const float bb = bias[c];
#pragma unroll
    for (int e = 0; e < 8; ++e) {
      const int R = m0 + e + 8 * hi;
      const int b = R >> 12, s = R & 4095;
      const float val = acc[t][e] + bb;
      const size_t idx = (((size_t)(b * Hc + h)) * Sc + s) * Dc + d;
      if (which == 0)      qbf[idx] = f2bf(val * scale);
      else if (which == 1) kbf[idx] = f2bf(val * scale);
      else {
        vf[idx] = val;
        vT[(((size_t)(b * Hc + h)) * Dc + d) * Sc + s] = f2bf(val);
      }
    }
  }
}

// ---------------------------------------------------------------------------
// 3) Landmarks: mean over 64-wide segments.  x:[BH,S,D] -> xl:[BH,L,D]
// ---------------------------------------------------------------------------
__global__ void k_landmarks(const uint16_t* __restrict__ x,
                            uint16_t* __restrict__ xl) {
  const int idx = blockIdx.x * blockDim.x + threadIdx.x;  // BH*L*D = 262144
  if (idx >= BH * Lc * Dc) return;
  const int d = idx & 63;
  const int l = (idx >> 6) & 63;
  const int bh = idx >> 12;
  const uint16_t* base = x + ((size_t)bh * Sc + l * 64) * Dc + d;
  float s = 0.f;
  for (int j = 0; j < 64; ++j) s += bf2f(base[(size_t)j * Dc]);
  xl[idx] = f2bf(s * (1.0f / 64.0f));
}

// ---------------------------------------------------------------------------
// 4) kernel_1 = softmax(q @ k_l^T) fused (rows of 64 live in one wave)
//    out probs1 bf16 [BH,S,L]
// ---------------------------------------------------------------------------
__global__ __launch_bounds__(32)
void k_kernel1(const uint16_t* __restrict__ qbf, const uint16_t* __restrict__ klbf,
               uint16_t* __restrict__ probs1) {
  const int bh = blockIdx.y;
  const int m0 = blockIdx.x * 16;
  const int lane = threadIdx.x & 31;
  const uint16_t* A  = qbf  + (size_t)bh * Sc * Dc;
  const uint16_t* Bm = klbf + (size_t)bh * Lc * Dc;

  f32x8 acc[4] = {};
#pragma unroll
  for (int k0 = 0; k0 < Dc; k0 += 32) {
    bf16x16 a = load_a_frag(A, Dc, m0, k0);
#pragma unroll
    for (int t = 0; t < 4; ++t)
      acc[t] = wmma_bf(a, load_b_frag(Bm, Dc, 16 * t, k0), acc[t]);
  }

  const int hi = lane >> 4;
#pragma unroll
  for (int e = 0; e < 8; ++e) {
    float m = fmaxf(fmaxf(acc[0][e], acc[1][e]), fmaxf(acc[2][e], acc[3][e]));
    m = fmaxf(m, __shfl_xor(m, 1, 32));
    m = fmaxf(m, __shfl_xor(m, 2, 32));
    m = fmaxf(m, __shfl_xor(m, 4, 32));
    m = fmaxf(m, __shfl_xor(m, 8, 32));
    float sum = 0.f;
#pragma unroll
    for (int t = 0; t < 4; ++t) { acc[t][e] = __expf(acc[t][e] - m); sum += acc[t][e]; }
    sum += __shfl_xor(sum, 1, 32);
    sum += __shfl_xor(sum, 2, 32);
    sum += __shfl_xor(sum, 4, 32);
    sum += __shfl_xor(sum, 8, 32);
    const float inv = 1.0f / sum;
    const int s = m0 + e + 8 * hi;
#pragma unroll
    for (int t = 0; t < 4; ++t)
      probs1[((size_t)bh * Sc + s) * Lc + 16 * t + (lane & 15)] = f2bf(acc[t][e] * inv);
  }
}

// ---------------------------------------------------------------------------
// 5) kernel_3 raw scores: q_l @ k^T  -> scores f32 [BH,L,S]
// ---------------------------------------------------------------------------
__global__ __launch_bounds__(32)
void k_scores3(const uint16_t* __restrict__ qlbf, const uint16_t* __restrict__ kbf,
               float* __restrict__ scores) {
  const int bh = blockIdx.z;
  const int m0 = blockIdx.y * 16;        // L tile
  const int n0 = blockIdx.x * 64;        // S group
  const int lane = threadIdx.x & 31;
  const uint16_t* A  = qlbf + (size_t)bh * Lc * Dc;
  const uint16_t* Bm = kbf  + (size_t)bh * Sc * Dc;  // N-major: row s, contiguous d

  f32x8 acc[4] = {};
#pragma unroll
  for (int k0 = 0; k0 < Dc; k0 += 32) {
    bf16x16 a = load_a_frag(A, Dc, m0, k0);
#pragma unroll
    for (int t = 0; t < 4; ++t)
      acc[t] = wmma_bf(a, load_b_frag(Bm, Dc, n0 + 16 * t, k0), acc[t]);
  }
  const int hi = lane >> 4;
#pragma unroll
  for (int t = 0; t < 4; ++t)
#pragma unroll
    for (int e = 0; e < 8; ++e) {
      const int l = m0 + e + 8 * hi;
      const int s = n0 + 16 * t + (lane & 15);
      scores[((size_t)bh * Lc + l) * Sc + s] = acc[t][e];
    }
}

// ---------------------------------------------------------------------------
// 6) row softmax over length-4096 rows -> probs3 bf16 [BH,L,S]
// ---------------------------------------------------------------------------
__global__ __launch_bounds__(256)
void k_softmax3(const float* __restrict__ scores, uint16_t* __restrict__ probs3) {
  const int row = blockIdx.x;            // BH*L = 4096
  const int tid = threadIdx.x;
  const float* src = scores + (size_t)row * Sc;
  float vals[16];
  float m = -3.4e38f;
#pragma unroll
  for (int i = 0; i < 16; ++i) { vals[i] = src[tid + 256 * i]; m = fmaxf(m, vals[i]); }
  m = fmaxf(m, __shfl_xor(m, 1, 32));
  m = fmaxf(m, __shfl_xor(m, 2, 32));
  m = fmaxf(m, __shfl_xor(m, 4, 32));
  m = fmaxf(m, __shfl_xor(m, 8, 32));
  m = fmaxf(m, __shfl_xor(m, 16, 32));
  __shared__ float red[8];
  const int wid = tid >> 5, lane = tid & 31;
  if (lane == 0) red[wid] = m;
  __syncthreads();
  if (tid == 0) {
    float mm = red[0];
    for (int i = 1; i < 8; ++i) mm = fmaxf(mm, red[i]);
    red[0] = mm;
  }
  __syncthreads();
  m = red[0];
  __syncthreads();
  float s = 0.f;
#pragma unroll
  for (int i = 0; i < 16; ++i) { vals[i] = __expf(vals[i] - m); s += vals[i]; }
  s += __shfl_xor(s, 1, 32);
  s += __shfl_xor(s, 2, 32);
  s += __shfl_xor(s, 4, 32);
  s += __shfl_xor(s, 8, 32);
  s += __shfl_xor(s, 16, 32);
  if (lane == 0) red[wid] = s;
  __syncthreads();
  if (tid == 0) {
    float ss = 0.f;
    for (int i = 0; i < 8; ++i) ss += red[i];
    red[0] = ss;
  }
  __syncthreads();
  const float inv = 1.0f / red[0];
  uint16_t* dst = probs3 + (size_t)row * Sc;
#pragma unroll
  for (int i = 0; i < 16; ++i) dst[tid + 256 * i] = f2bf(vals[i] * inv);
}

// ---------------------------------------------------------------------------
// 7) new_value = probs3 @ v  -> nv f32 [BH,L,D]   (B from vT, N-major)
// ---------------------------------------------------------------------------
__global__ __launch_bounds__(32)
void k_newvalue(const uint16_t* __restrict__ probs3, const uint16_t* __restrict__ vT,
                float* __restrict__ nv) {
  const int bh = blockIdx.y;
  const int m0 = blockIdx.x * 16;       // L tile
  const int lane = threadIdx.x & 31;
  const uint16_t* A  = probs3 + (size_t)bh * Lc * Sc;   // lda = S
  const uint16_t* Bm = vT     + (size_t)bh * Dc * Sc;   // row d, contiguous s

  f32x8 acc[4] = {};
  for (int k0 = 0; k0 < Sc; k0 += 32) {
    bf16x16 a = load_a_frag(A, Sc, m0, k0);
#pragma unroll
    for (int t = 0; t < 4; ++t)
      acc[t] = wmma_bf(a, load_b_frag(Bm, Sc, 16 * t, k0), acc[t]);
  }
  const int hi = lane >> 4;
#pragma unroll
  for (int t = 0; t < 4; ++t)
#pragma unroll
    for (int e = 0; e < 8; ++e) {
      const int l = m0 + e + 8 * hi;
      const int d = 16 * t + (lane & 15);
      nv[((size_t)bh * Lc + l) * Dc + d] = acc[t][e];
    }
}

// ---------------------------------------------------------------------------
// 8) kernel_2 = softmax(q_l @ k_l^T) in f32, + per-head max column sum
// ---------------------------------------------------------------------------
__global__ __launch_bounds__(256)
void k_kernel2(const uint16_t* __restrict__ qlbf, const uint16_t* __restrict__ klbf,
               float* __restrict__ K2, float* __restrict__ maxv) {
  const int bh = blockIdx.x;
  const int tid = threadIdx.x;
  __shared__ float Ql[Lc * Dc], Kl[Lc * Dc], G[Lc * Lc];
  __shared__ float cs[Lc];
  for (int i = tid; i < Lc * Dc; i += 256) {
    Ql[i] = bf2f(qlbf[(size_t)bh * Lc * Dc + i]);
    Kl[i] = bf2f(klbf[(size_t)bh * Lc * Dc + i]);
  }
  __syncthreads();
  for (int idx = tid; idx < Lc * Lc; idx += 256) {
    const int i = idx >> 6, j = idx & 63;
    float s = 0.f;
    for (int d = 0; d < Dc; ++d) s += Ql[i * Dc + d] * Kl[j * Dc + d];
    G[idx] = s;
  }
  __syncthreads();
  if (tid < Lc) {                         // one thread per row softmax
    float* row = G + tid * Lc;
    float m = row[0];
    for (int j = 1; j < Lc; ++j) m = fmaxf(m, row[j]);
    float sum = 0.f;
    for (int j = 0; j < Lc; ++j) { row[j] = __expf(row[j] - m); sum += row[j]; }
    const float inv = 1.0f / sum;
    for (int j = 0; j < Lc; ++j) row[j] *= inv;
  }
  __syncthreads();
  for (int idx = tid; idx < Lc * Lc; idx += 256)
    K2[(size_t)bh * Lc * Lc + idx] = G[idx];
  if (tid < Lc) {                         // column sums
    float s = 0.f;
    for (int i = 0; i < Lc; ++i) s += G[i * Lc + tid];
    cs[tid] = s;
  }
  __syncthreads();
  if (tid == 0) {
    float m = cs[0];
    for (int j = 1; j < Lc; ++j) m = fmaxf(m, cs[j]);
    maxv[bh] = m;
  }
}

// ---------------------------------------------------------------------------
// 9) iterative_inv (6 Newton iters) then M2 = inv @ new_value; write M2^T bf16
// ---------------------------------------------------------------------------
__device__ __forceinline__ void mm64(const float* __restrict__ A,
                                     const float* __restrict__ B,
                                     float* __restrict__ C, int tid, float scale) {
  float r[16];
#pragma unroll
  for (int t = 0; t < 16; ++t) {
    const int idx = tid + 256 * t;
    const int i = idx >> 6, j = idx & 63;
    float s = 0.f;
    for (int k = 0; k < 64; ++k) s += A[i * 64 + k] * B[k * 64 + j];
    r[t] = scale * s;
  }
  __syncthreads();
#pragma unroll
  for (int t = 0; t < 16; ++t) C[tid + 256 * t] = r[t];
  __syncthreads();
}

__global__ __launch_bounds__(256)
void k_inv(const float* __restrict__ K2, const float* __restrict__ maxv,
           const float* __restrict__ nv, uint16_t* __restrict__ M2T) {
  const int bh = blockIdx.x;
  const int tid = threadIdx.x;
  __shared__ float Key[4096], V[4096], T1[4096], T2[4096];

  // global max over all heads (reference: jnp.max over whole array)
  float gm = maxv[0];
  for (int i = 1; i < BH; ++i) gm = fmaxf(gm, maxv[i]);
  const float sc = 1.0f / gm;

  for (int t = 0; t < 16; ++t)
    Key[tid + 256 * t] = K2[(size_t)bh * 4096 + tid + 256 * t];
  __syncthreads();
  for (int t = 0; t < 16; ++t) {
    const int idx = tid + 256 * t;
    const int i = idx >> 6, j = idx & 63;
    V[idx] = sc * Key[j * 64 + i];
  }
  __syncthreads();

  for (int it = 0; it < 6; ++it) {
    mm64(Key, V, T1, tid, 1.0f);          // kv
    mm64(T1, T1, T2, tid, 1.0f);          // kv^2
    for (int t = 0; t < 16; ++t) {        // U = 15I - 7kv + kv^2
      const int idx = tid + 256 * t;
      const int i = idx >> 6, j = idx & 63;
      T2[idx] = ((i == j) ? 15.f : 0.f) - 7.f * T1[idx] + T2[idx];
    }
    __syncthreads();
    mm64(T1, T2, T1, tid, 1.0f);          // W = kv @ U (reg-staged, alias-safe)
    for (int t = 0; t < 16; ++t) {        // F = 13I - W
      const int idx = tid + 256 * t;
      const int i = idx >> 6, j = idx & 63;
      T1[idx] = ((i == j) ? 13.f : 0.f) - T1[idx];
    }
    __syncthreads();
    mm64(V, T1, V, tid, 0.25f);           // V = 0.25 * V @ F
  }

  for (int t = 0; t < 16; ++t)            // T1 = new_value [L,D]
    T1[tid + 256 * t] = nv[(size_t)bh * 4096 + tid + 256 * t];
  __syncthreads();
  for (int t = 0; t < 16; ++t) {          // M2 = V @ nv ; store transposed bf16
    const int idx = tid + 256 * t;
    const int i = idx >> 6, j = idx & 63;
    float s = 0.f;
    for (int k = 0; k < 64; ++k) s += V[i * 64 + k] * T1[k * 64 + j];
    M2T[((size_t)bh * 64 + j) * 64 + i] = f2bf(s);
  }
}

// ---------------------------------------------------------------------------
// 10) depthwise conv over sequence (K=65) -> writes out[] (context added later)
// ---------------------------------------------------------------------------
__global__ __launch_bounds__(256)
void k_conv(const float* __restrict__ vf, const float* __restrict__ cw,
            float* __restrict__ out) {
  const int bh = blockIdx.y;
  const int s0 = blockIdx.x * 64;
  const int b = bh >> 4, h = bh & 15;
  __shared__ float tile[128 * 64];
  __shared__ float w[Kc];
  const int tid = threadIdx.x;
  if (tid < Kc) w[tid] = cw[h * Kc + tid];
  const float* src = vf + (size_t)bh * Sc * Dc;
  for (int i = tid; i < 128 * 64; i += 256) {
    const int r = i >> 6, d = i & 63;
    const int s = s0 - 32 + r;
    tile[i] = (s >= 0 && s < Sc) ? src[(size_t)s * Dc + d] : 0.f;
  }
  __syncthreads();
  const int d = tid & 63;
  for (int rr = (tid >> 6); rr < 64; rr += 4) {
    float s = 0.f;
#pragma unroll
    for (int t = 0; t < Kc; ++t) s += w[t] * tile[(rr + t) * 64 + d];
    out[((size_t)(b * Sc + s0 + rr)) * HIDc + h * Dc + d] = s;
  }
}

// ---------------------------------------------------------------------------
// 11) context = probs1 @ M2, accumulate onto conv output in [B,S,HID]
// ---------------------------------------------------------------------------
__global__ __launch_bounds__(32)
void k_context(const uint16_t* __restrict__ probs1, const uint16_t* __restrict__ M2T,
               float* __restrict__ out) {
  const int bh = blockIdx.y;
  const int m0 = blockIdx.x * 16;
  const int lane = threadIdx.x & 31;
  const uint16_t* A  = probs1 + (size_t)bh * Sc * Lc;   // lda = L
  const uint16_t* Bm = M2T    + (size_t)bh * Dc * Lc;   // row d, contiguous l

  f32x8 acc[4] = {};
#pragma unroll
  for (int k0 = 0; k0 < Lc; k0 += 32) {
    bf16x16 a = load_a_frag(A, Lc, m0, k0);
#pragma unroll
    for (int t = 0; t < 4; ++t)
      acc[t] = wmma_bf(a, load_b_frag(Bm, Lc, 16 * t, k0), acc[t]);
  }
  const int b = bh >> 4, h = bh & 15, hi = lane >> 4;
#pragma unroll
  for (int t = 0; t < 4; ++t)
#pragma unroll
    for (int e = 0; e < 8; ++e) {
      const int s = m0 + e + 8 * hi;
      const int d = 16 * t + (lane & 15);
      const size_t idx = ((size_t)(b * Sc + s)) * HIDc + h * Dc + d;
      out[idx] += acc[t][e];
    }
}

// ---------------------------------------------------------------------------
// host launcher
// ---------------------------------------------------------------------------
extern "C" void kernel_launch(void* const* d_in, const int* in_sizes, int n_in,
                              void* d_out, int out_size, void* d_ws, size_t ws_size,
                              hipStream_t stream) {
  const float* hs   = (const float*)d_in[0];
  const float* Wq   = (const float*)d_in[1];
  const float* bq   = (const float*)d_in[2];
  const float* Wk   = (const float*)d_in[3];
  const float* bk   = (const float*)d_in[4];
  const float* Wv   = (const float*)d_in[5];
  const float* bv   = (const float*)d_in[6];
  const float* cw   = (const float*)d_in[7];
  float* out = (float*)d_out;
  (void)in_sizes; (void)n_in; (void)out_size; (void)ws_size;

  // bump allocator over workspace (~335 MB needed)
  char* wp = (char*)d_ws;
  auto take = [&](size_t bytes) -> void* {
    void* r = (void*)wp;
    wp += (bytes + 255) & ~(size_t)255;
    return r;
  };
  uint16_t* hs_bf  = (uint16_t*)take((size_t)ROWS * HIDc * 2);
  uint16_t* wq_bf  = (uint16_t*)take((size_t)HIDc * HIDc * 2);
  uint16_t* wk_bf  = (uint16_t*)take((size_t)HIDc * HIDc * 2);
  uint16_t* wv_bf  = (uint16_t*)take((size_t)HIDc * HIDc * 2);
  uint16_t* q_bf   = (uint16_t*)take((size_t)BH * Sc * Dc * 2);
  uint16_t* k_bf   = (uint16_t*)take((size_t)BH * Sc * Dc * 2);
  float*    v_f32  = (float*)   take((size_t)BH * Sc * Dc * 4);
  uint16_t* vT_bf  = (uint16_t*)take((size_t)BH * Dc * Sc * 2);
  uint16_t* ql_bf  = (uint16_t*)take((size_t)BH * Lc * Dc * 2);
  uint16_t* kl_bf  = (uint16_t*)take((size_t)BH * Lc * Dc * 2);
  uint16_t* probs1 = (uint16_t*)take((size_t)BH * Sc * Lc * 2);
  float*    sc3    = (float*)   take((size_t)BH * Lc * Sc * 4);
  uint16_t* probs3 = (uint16_t*)take((size_t)BH * Lc * Sc * 2);
  float*    nv     = (float*)   take((size_t)BH * Lc * Dc * 4);
  float*    K2     = (float*)   take((size_t)BH * Lc * Lc * 4);
  float*    maxv   = (float*)   take((size_t)BH * 4);
  uint16_t* M2T    = (uint16_t*)take((size_t)BH * Dc * Lc * 2);

  const int nHS = ROWS * HIDc;        // 16777216
  const int nW  = HIDc * HIDc;        // 1048576
  k_f32_to_bf16<<<dim3((nHS + 255) / 256), dim3(256), 0, stream>>>(hs, hs_bf, nHS);
  k_f32_to_bf16<<<dim3((nW + 255) / 256), dim3(256), 0, stream>>>(Wq, wq_bf, nW);
  k_f32_to_bf16<<<dim3((nW + 255) / 256), dim3(256), 0, stream>>>(Wk, wk_bf, nW);
  k_f32_to_bf16<<<dim3((nW + 255) / 256), dim3(256), 0, stream>>>(Wv, wv_bf, nW);

  k_qkv_gemm<<<dim3(ROWS / 16, 2, 3), dim3(256), 0, stream>>>(
      hs_bf, wq_bf, wk_bf, wv_bf, bq, bk, bv, q_bf, k_bf, v_f32, vT_bf);

  k_landmarks<<<dim3(BH * Lc * Dc / 256), dim3(256), 0, stream>>>(q_bf, ql_bf);
  k_landmarks<<<dim3(BH * Lc * Dc / 256), dim3(256), 0, stream>>>(k_bf, kl_bf);

  k_kernel1<<<dim3(Sc / 16, BH), dim3(32), 0, stream>>>(q_bf, kl_bf, probs1);

  k_scores3<<<dim3(Sc / 64, Lc / 16, BH), dim3(32), 0, stream>>>(ql_bf, k_bf, sc3);
  k_softmax3<<<dim3(BH * Lc), dim3(256), 0, stream>>>(sc3, probs3);
  k_newvalue<<<dim3(Lc / 16, BH), dim3(32), 0, stream>>>(probs3, vT_bf, nv);

  k_kernel2<<<dim3(BH), dim3(256), 0, stream>>>(ql_bf, kl_bf, K2, maxv);
  k_inv<<<dim3(BH), dim3(256), 0, stream>>>(K2, maxv, nv, M2T);

  k_conv<<<dim3(Sc / 64, BH), dim3(256), 0, stream>>>(v_f32, cw, out);
  k_context<<<dim3(Sc / 16, BH), dim3(32), 0, stream>>>(probs1, M2T, out);
}
